// baseline_snn_54760833024459
// MI455X (gfx1250) — compile-verified, matrix operand order checked by reading, not run on hardware
//
#include <hip/hip_runtime.h>
#include <hip/hip_bf16.h>

// ---------------- types ----------------
typedef _Float16 f16;
typedef _Float16 v16h __attribute__((ext_vector_type(16)));
typedef _Float16 v8h  __attribute__((ext_vector_type(8)));
typedef float    v8f  __attribute__((ext_vector_type(8)));

// ---------------- SNN constants (tau_m=4, tau_s=1) ----------------
#define A1F   1.1466802242428472f     // e^-1/4 + e^-1
#define A2F  (-0.2865047968601901f)   // -e^(-1/4-1)
#define V0F   2.1165347359575992f     // 4^(4/3)/3
#define SIGMAF 0.7788007830714049f    // e^-1/4
#define TLEN  100
#define BATCH 256

// ---------------- workspace layout (bytes) ----------------
// total required: ~77.2 MB (entire working set fits in the 192MB L2)
#define OFF_C     ((size_t)0)                         // c[t], 100 f32
#define OFF_IN16  ((size_t)1024)                      // inputs f16 [256][800]
#define OFF_W1F   (OFF_IN16 + (size_t)256*800*2)      // w1 f16 [512][800]
#define OFF_G1    (OFF_W1F  + (size_t)512*800*2)      // inputs@w1^T f32 [256][512]
#define OFF_W2F   (OFF_G1   + (size_t)256*512*4)      // w2 f16 [512][512]
#define OFF_W3F   (OFF_W2F  + (size_t)512*512*2)      // w3 f16 [16][512]
#define OFF_H2    (OFF_W3F  + (size_t)16*512*2)       // h2 f16 [25600][512]; reused as h3
#define OFF_CUR2  (OFF_H2   + (size_t)25600*512*2)    // cur2 f32 [25600][512]
#define OFF_CUR3  OFF_IN16                            // cur3 f32 [25600][16] (reuses dead in16/w1/G1)

// ---------------- c[t] scalar IIR (input is constant 1 after factoring) ----------------
__global__ void k_coeffs(float* __restrict__ c) {
    if (blockIdx.x == 0 && threadIdx.x == 0) {
        float s1 = 0.f, s2 = 0.f;
        for (int t = 0; t < TLEN; ++t) {
            float y = A1F * s1 + A2F * s2 + V0F;
            c[t] = y; s2 = s1; s1 = y;
        }
    }
}

// ---------------- f32 -> f16 convert with zero padding ----------------
__global__ void k_cvt_pad(const float* __restrict__ src, int R, int C,
                          f16* __restrict__ dst, int Cp, int total) {
    int i = blockIdx.x * blockDim.x + threadIdx.x;
    if (i >= total) return;
    int r = i / Cp, cc = i - r * Cp;
    float v = (r < R && cc < C) ? src[(size_t)r * C + cc] : 0.0f;
    dst[i] = (f16)v;
}

// ---- WMMA GEMM: C[M,Np] = A[M,Kp] * B[Np,Kp]^T, f16 in / f32 accum ------------------
// A, B stored K-contiguous (row-major), rows zero-padded to Kp (multiple of 32 halfs).
// Per the CDNA5 16-bit 16x32 fragment layout, lane l (g=l>>4, m=l&15) holds row m,
// K = {8g..8g+7} and {16+8g..16+8g+7}: two contiguous 16B chunks -> two b128 loads.
// Each wave computes a 16 x (16*NT) tile: one A fragment feeds NT independent WMMA
// accumulator chains (cuts A-side L2 traffic NT-fold; breaks WMMA->WMMA RAW chains).
template <int NT>
__global__ void k_gemm_f16(const f16* __restrict__ A, const f16* __restrict__ B,
                           float* __restrict__ C, int Kp, int Np, int Ksteps) {
    const int lane = threadIdx.x;          // wave32, one wave per block
    const int g    = lane >> 4;
    const int mn   = lane & 15;
    const int nt0  = blockIdx.x * NT;
    const int mtile = blockIdx.y;

    const f16* arow = A + ((size_t)(mtile * 16 + mn)) * Kp + 8 * g;
    const f16* brow[NT];
#pragma unroll
    for (int j = 0; j < NT; ++j)
        brow[j] = B + ((size_t)((nt0 + j) * 16 + mn)) * Kp + 8 * g;

    v8f acc[NT];
#pragma unroll
    for (int j = 0; j < NT; ++j) acc[j] = (v8f){0.f,0.f,0.f,0.f,0.f,0.f,0.f,0.f};

    for (int ks = 0; ks < Ksteps; ++ks) {
        const f16* ap = arow + ks * 32;
        __builtin_prefetch(ap + 4 * 32, 0, 0);          // global_prefetch_b8
        v8h a0 = *(const v8h*)(ap);
        v8h a1 = *(const v8h*)(ap + 16);
        v16h av = __builtin_shufflevector(a0, a1, 0,1,2,3,4,5,6,7,8,9,10,11,12,13,14,15);
#pragma unroll
        for (int j = 0; j < NT; ++j) {
            const f16* bp = brow[j] + ks * 32;
            __builtin_prefetch(bp + 4 * 32, 0, 0);
            v8h b0 = *(const v8h*)(bp);
            v8h b1 = *(const v8h*)(bp + 16);
            v16h bv = __builtin_shufflevector(b0, b1, 0,1,2,3,4,5,6,7,8,9,10,11,12,13,14,15);
            acc[j] = __builtin_amdgcn_wmma_f32_16x16x32_f16(
                /*neg_a=*/false, av, /*neg_b=*/false, bv,
                /*c_mod=*/(short)0, acc[j], /*reuse_a=*/false, /*reuse_b=*/false);
        }
    }
    // C/D layout: lane holds N = ntile*16 + mn, rows M = 8*g + d  (d = VGPR index 0..7)
#pragma unroll
    for (int j = 0; j < NT; ++j) {
        float* crow = C + ((size_t)(mtile * 16 + 8 * g)) * Np + (size_t)(nt0 + j) * 16 + mn;
#pragma unroll
        for (int d = 0; d < 8; ++d) crow[(size_t)d * Np] = acc[j][d];
    }
}

// ---------------- fused LIF layer-1 scan (cur1[t]=c[t]*G1+b1) + IIR for layer 2 ------
__global__ void k_lif1_iir2(const float* __restrict__ G1, const float* __restrict__ b1,
                            const float* __restrict__ c, f16* __restrict__ h2) {
    int idx = blockIdx.x * blockDim.x + threadIdx.x;   // 256 * 512
    int o = idx & 511;
    float g    = G1[idx];
    float bias = (o < 500) ? b1[o] : 0.0f;
    float v = 0.f, s = 0.f, y1 = 0.f, y2 = 0.f;
    f16* out = h2 + (size_t)idx;
    for (int t = 0; t < TLEN; ++t) {
        float cur = c[t] * g + bias;
        v = SIGMAF * v * (1.0f - s) + cur;
        s = (v > 1.0f) ? 1.0f : 0.0f;
        float y = A1F * y1 + A2F * y2 + V0F * s;
        y2 = y1; y1 = y;
        out[(size_t)t * (BATCH * 512)] = (f16)y;
    }
}

// ---------------- fused LIF layer-2 scan + IIR for layer 3 ----------------
__global__ void k_lif2_iir3(const float* __restrict__ cur2, const float* __restrict__ b2,
                            f16* __restrict__ h3) {
    int idx = blockIdx.x * blockDim.x + threadIdx.x;   // 256 * 512
    int o = idx & 511;
    float bias = (o < 500) ? b2[o] : 0.0f;
    const float* cp = cur2 + (size_t)idx;
    f16* out = h3 + (size_t)idx;
    float v = 0.f, s = 0.f, y1 = 0.f, y2 = 0.f;
    for (int t = 0; t < TLEN; ++t) {
        float cur = cp[(size_t)t * (BATCH * 512)] + bias;
        v = SIGMAF * v * (1.0f - s) + cur;
        s = (v > 1.0f) ? 1.0f : 0.0f;
        float y = A1F * y1 + A2F * y2 + V0F * s;
        y2 = y1; y1 = y;
        out[(size_t)t * (BATCH * 512)] = (f16)y;
    }
}

// ---------------- LIF layer-3 scan -> output spikes [B,10,T] f32 ----------------
__global__ void k_lif3(const float* __restrict__ cur3, const float* __restrict__ b3,
                       float* __restrict__ out) {
    int idx = blockIdx.x * blockDim.x + threadIdx.x;   // 256 * 10
    if (idx >= BATCH * 10) return;
    int b = idx / 10, o = idx - b * 10;
    float bias = b3[o];
    const float* cp = cur3 + (size_t)b * 16 + o;       // cur3 row = t*256+b, pitch 16
    float* op = out + (size_t)idx * TLEN;              // [b][o][t]
    float v = 0.f, s = 0.f;
    for (int t = 0; t < TLEN; ++t) {
        v = SIGMAF * v * (1.0f - s) + cp[(size_t)t * (BATCH * 16)] + bias;
        s = (v > 1.0f) ? 1.0f : 0.0f;
        op[t] = s;
    }
}

extern "C" void kernel_launch(void* const* d_in, const int* in_sizes, int n_in,
                              void* d_out, int out_size, void* d_ws, size_t ws_size,
                              hipStream_t stream) {
    const float* inputs = (const float*)d_in[0];   // [256,784]
    const float* w1     = (const float*)d_in[1];   // [500,784]
    const float* b1     = (const float*)d_in[2];   // [500]
    const float* w2     = (const float*)d_in[3];   // [500,500]
    const float* b2     = (const float*)d_in[4];   // [500]
    const float* w3     = (const float*)d_in[5];   // [10,500]
    const float* b3     = (const float*)d_in[6];   // [10]
    float* out = (float*)d_out;                    // [256,10,100] f32
    char* ws = (char*)d_ws;

    float* c    = (float*)(ws + OFF_C);
    f16*   in16 = (f16*)  (ws + OFF_IN16);
    f16*   w1f  = (f16*)  (ws + OFF_W1F);
    float* G1   = (float*)(ws + OFF_G1);
    f16*   w2f  = (f16*)  (ws + OFF_W2F);
    f16*   w3f  = (f16*)  (ws + OFF_W3F);
    f16*   h2   = (f16*)  (ws + OFF_H2);       // also reused as h3
    float* cur2 = (float*)(ws + OFF_CUR2);
    float* cur3 = (float*)(ws + OFF_CUR3);

    // 1) layer-1 time coefficients c[t]
    k_coeffs<<<1, 32, 0, stream>>>(c);

    // 2) f16 conversions with zero padding (K padded to mult of 32, N to mult of 16)
    k_cvt_pad<<<(256 * 800 + 255) / 256, 256, 0, stream>>>(inputs, 256, 784, in16, 800, 256 * 800);
    k_cvt_pad<<<(512 * 800 + 255) / 256, 256, 0, stream>>>(w1, 500, 784, w1f, 800, 512 * 800);
    k_cvt_pad<<<(512 * 512 + 255) / 256, 256, 0, stream>>>(w2, 500, 500, w2f, 512, 512 * 512);
    k_cvt_pad<<<(16 * 512 + 255) / 256, 256, 0, stream>>>(w3, 10, 500, w3f, 512, 16 * 512);

    // 3) G1 = inputs @ w1^T : [256,800]x[512,800]^T -> [256,512], K-steps = 25
    k_gemm_f16<4><<<dim3(8, 16), 32, 0, stream>>>(in16, w1f, G1, 800, 512, 25);

    // 4) LIF layer 1 (cur1[t] = c[t]*G1 + b1) fused with IIR -> h2 f16 [25600][512]
    k_lif1_iir2<<<(BATCH * 512) / 256, 256, 0, stream>>>(G1, b1, c, h2);

    // 5) cur2 = h2 @ w2^T : [25600,512]x[512,512]^T, K-steps = 16 (dominant GEMM)
    k_gemm_f16<4><<<dim3(8, 1600), 32, 0, stream>>>(h2, w2f, cur2, 512, 512, 16);

    // 6) LIF layer 2 fused with IIR -> h3 (reuses h2 storage)
    k_lif2_iir3<<<(BATCH * 512) / 256, 256, 0, stream>>>(cur2, b2, h2);

    // 7) cur3 = h3 @ w3^T : [25600,512]x[16,512]^T -> [25600,16]
    k_gemm_f16<1><<<dim3(1, 1600), 32, 0, stream>>>(h2, w3f, cur3, 512, 16, 16);

    // 8) LIF layer 3 -> spikes, output layout [B,10,T]
    k_lif3<<<(BATCH * 10 + 255) / 256, 256, 0, stream>>>(cur3, b3, out);
}
